// TransformerBlock_45792941309983
// MI455X (gfx1250) — matrix operand order repaired and checked by reading, not compiled, not run on hardware
//
#include <hip/hip_runtime.h>
#include <hip/hip_bf16.h>
#include <math.h>
#include <stdint.h>

// ---------------------------------------------------------------------------
// Types / helpers
// ---------------------------------------------------------------------------
typedef __attribute__((ext_vector_type(16))) __bf16 v16bf;
typedef __attribute__((ext_vector_type(8)))  float  v8f;

union FragBF { uint4 u[2]; v16bf v; };

__device__ __forceinline__ unsigned short f2bf(float f) {
  unsigned int u = __float_as_uint(f);
  u += 0x7FFFu + ((u >> 16) & 1u);          // round-to-nearest-even
  return (unsigned short)(u >> 16);
}
__device__ __forceinline__ float bf2f(unsigned short h) {
  return __uint_as_float(((unsigned int)h) << 16);
}

// --- CDNA5 async global->LDS copies (ASYNCcnt-tracked DMA, no VGPR bounce).
// INST_OFFSET is added to BOTH the LDS and global addresses (ISA 08 §4.4),
// so contiguous tile segments share one address pair + immediate offsets.
template <int OFF>
__device__ __forceinline__ void async_g2l_b128o(unsigned lds, unsigned goff,
                                                unsigned long long gbase) {
  asm volatile("global_load_async_to_lds_b128 %0, %1, %2 offset:%3"
               :: "v"(lds), "v"(goff), "s"(gbase), "n"(OFF) : "memory");
}
__device__ __forceinline__ void async_g2l_b64(unsigned lds, unsigned goff,
                                              unsigned long long gbase) {
  asm volatile("global_load_async_to_lds_b64 %0, %1, %2"
               :: "v"(lds), "v"(goff), "s"(gbase) : "memory");
}
template <int N>
__device__ __forceinline__ void wait_asynccnt() {
  asm volatile("s_wait_asynccnt %0" :: "n"(N) : "memory");
}
__device__ __forceinline__ unsigned lds_addr(const void* p) {
  return (unsigned)(uintptr_t)p;   // low 32 bits of flat shared address = LDS offset
}

// A-matrix fragment (16x32 bf16): per lane (row = lane&15, sel = lane>>4):
// K = sel*8..+7 and K = sel*8+16..+23
__device__ __forceinline__ v16bf load_a_frag(const unsigned short* base, int stride, int lane) {
  const int r = lane & 15, sel = lane >> 4;
  const unsigned short* p = base + r * stride + sel * 8;
  FragBF f;
  f.u[0] = *(const uint4*)(p);
  f.u[1] = *(const uint4*)(p + 16);
  return f.v;
}
// B-matrix fragment (32x16 bf16), LDS stored transposed [n][k]:
// lane n = lane&15, sel = lane>>4: K = sel*8..+7 and K = sel*8+16..+23
__device__ __forceinline__ v16bf load_b_frag(const unsigned short* base, int stride, int lane) {
  const int n = lane & 15, sel = lane >> 4;
  const unsigned short* p = base + n * stride + sel * 8;
  FragBF f;
  f.u[0] = *(const uint4*)(p);
  f.u[1] = *(const uint4*)(p + 16);
  return f.v;
}
__device__ __forceinline__ v8f wmma_bf16(v16bf a, v16bf b, v8f c) {
  return __builtin_amdgcn_wmma_f32_16x16x32_bf16(false, a, false, b, (short)0, c, false, false);
}

// ---------------------------------------------------------------------------
// fp32 [K][N] -> bf16 transposed [N][K] weight conversion (one-time, tiny)
// ---------------------------------------------------------------------------
__global__ __launch_bounds__(256)
void convert_transpose_kernel(const float* __restrict__ src, unsigned short* __restrict__ dst,
                              int K, int N) {
  const int i = blockIdx.x * 256 + threadIdx.x;
  if (i >= K * N) return;
  const int k = i / N, n = i % N;
  dst[(size_t)n * K + k] = f2bf(src[i]);
}

// ---------------------------------------------------------------------------
// Spiral-RoPE cos/sin tables (361 x 64), closed form
// ---------------------------------------------------------------------------
__global__ __launch_bounds__(256)
void rope_table_kernel(float* __restrict__ ropeC, float* __restrict__ ropeS) {
  int idx = blockIdx.x * 256 + threadIdx.x;
  if (idx >= 361 * 64) return;
  const int s = idx >> 6, d = idx & 63;
  const int k = d >> 4;
  const int kn = k & 1;
  const int tb = 2 * kn;
  const int ro = (d & 15) >> 1;
  const int to = (ro >> 1) * 4 + (ro & 1);
  int ti = tb + to; if (ti > 15) ti = 15;
  const float theta = powf(100.0f, -(float)ti / 16.0f);
  const float ang = (float)k * (3.14159265358979323846f / 4.0f);
  const float xf = (float)(s / 19), yf = (float)(s % 19);
  const float rot = theta * (xf * cosf(ang) + yf * sinf(ang));
  ropeC[idx] = cosf(rot);
  ropeS[idx] = sinf(rot);
}

// ---------------------------------------------------------------------------
// RMSNorm (wave32 per token) fp32 -> bf16
// ---------------------------------------------------------------------------
__global__ __launch_bounds__(256)
void rmsnorm_kernel(const float* __restrict__ x, const float* __restrict__ gamma,
                    unsigned short* __restrict__ out, int ntok) {
  const int wid = threadIdx.x >> 5, lane = threadIdx.x & 31;
  const int tok = blockIdx.x * 8 + wid;
  if (tok >= ntok) return;
  const float* row = x + (size_t)tok * 512;
  float ss = 0.f;
  for (int i = lane; i < 512; i += 32) { float v = row[i]; ss += v * v; }
  #pragma unroll
  for (int o = 16; o > 0; o >>= 1) ss += __shfl_xor(ss, o, 32);
  const float scale = rsqrtf(ss * (1.0f / 512.0f) + 1e-6f);
  unsigned short* orow = out + (size_t)tok * 512;
  for (int i = lane; i < 512; i += 32) orow[i] = f2bf(row[i] * scale * gamma[i]);
}

// ---------------------------------------------------------------------------
// 128x64-tile bf16 WMMA GEMM. 256 threads = 8 waves (4x2), wave tile 32x32
// (4 WMMAs per k-step; 8 for dual-B). Async global->LDS double buffering.
// B is pre-transposed: Bt[n][k].
// MODE 1: fp32 resid + acc.            MODE 2: dual-B silu(g)*u -> bf16.
// MODE 3: bf16 transposed Vt[b][n][384-padded s].
// MODE 4: bf16 with fused spiral-RoPE (partner column via __shfl_xor lane^1).
// ---------------------------------------------------------------------------
template <int MODE>
__global__ __launch_bounds__(256)
void gemm128_kernel(const unsigned short* __restrict__ A,
                    const unsigned short* __restrict__ Bt1,
                    const unsigned short* __restrict__ Bt2,
                    const float* __restrict__ resid,
                    const float* __restrict__ ropeC,
                    const float* __restrict__ ropeS,
                    void* __restrict__ outp,
                    int M, int N, int K) {
  constexpr int NB  = (MODE == 2) ? 2 : 1;
  constexpr int OPS = (MODE == 2) ? 4 : 3;   // async ops per thread per k-tile
  __shared__ __align__(16) unsigned short sA[2][128 * 32];
  __shared__ __align__(16) unsigned short sB[2][NB * 64 * 32];

  const int t = threadIdx.x;
  const int lane = t & 31, wid = t >> 5;
  const int m0 = blockIdx.y * 128, n0 = blockIdx.x * 64;
  const int waveM = wid >> 1, waveN = wid & 1;

  const unsigned long long aBase = (unsigned long long)(uintptr_t)A;
  const unsigned long long bBase1 = (unsigned long long)(uintptr_t)Bt1;
  const unsigned long long bBase2 = (unsigned long long)(uintptr_t)Bt2;

  // staging geometry (contiguous 16B segments in both LDS and global)
  const int ar = t >> 1, ac = (t & 1) * 16;   // A tile 128x32: 16 halfs/thread
  const int bn = t >> 2, bc = (t & 3) * 8;    // B tile 64x32 : 8 halfs/thread

  v8f acc00 = {}, acc01 = {}, acc10 = {}, acc11 = {};
  v8f accg00 = {}, accg01 = {}, accg10 = {}, accg11 = {};

  auto issue_tile = [&](int ke, int buf) {   // ke = k element offset
    const unsigned ldsA = lds_addr(&sA[buf][ar * 32 + ac]);
    const unsigned aoff = (unsigned)(((m0 + ar) * K + ke + ac) * 2);
    async_g2l_b128o<0>(ldsA, aoff, aBase);
    async_g2l_b128o<16>(ldsA, aoff, aBase);
    const unsigned ldsB = lds_addr(&sB[buf][bn * 32 + bc]);
    const unsigned boff = (unsigned)(((n0 + bn) * K + ke + bc) * 2);
    async_g2l_b128o<0>(ldsB, boff, bBase1);
    if (MODE == 2)
      async_g2l_b128o<0>(ldsB + 64 * 32 * 2, boff, bBase2);
  };

  issue_tile(0, 0);
  const int nk = K >> 5;
  for (int kt = 0; kt < nk; ++kt) {
    const int buf = kt & 1;
    if (kt + 1 < nk) { issue_tile((kt + 1) << 5, buf ^ 1); wait_asynccnt<OPS>(); }
    else             { wait_asynccnt<0>(); }
    __syncthreads();                       // tile kt resident for all waves

    const v16bf a0 = load_a_frag(&sA[buf][(waveM * 32) * 32],      32, lane);
    const v16bf a1 = load_a_frag(&sA[buf][(waveM * 32 + 16) * 32], 32, lane);
    const v16bf b0 = load_b_frag(&sB[buf][(waveN * 32) * 32],      32, lane);
    const v16bf b1 = load_b_frag(&sB[buf][(waveN * 32 + 16) * 32], 32, lane);
    acc00 = wmma_bf16(a0, b0, acc00);
    acc01 = wmma_bf16(a0, b1, acc01);
    acc10 = wmma_bf16(a1, b0, acc10);
    acc11 = wmma_bf16(a1, b1, acc11);
    if (MODE == 2) {
      const v16bf c0 = load_b_frag(&sB[buf][64 * 32 + (waveN * 32) * 32],      32, lane);
      const v16bf c1 = load_b_frag(&sB[buf][64 * 32 + (waveN * 32 + 16) * 32], 32, lane);
      accg00 = wmma_bf16(a0, c0, accg00);
      accg01 = wmma_bf16(a0, c1, accg01);
      accg10 = wmma_bf16(a1, c0, accg10);
      accg11 = wmma_bf16(a1, c1, accg11);
    }
    __syncthreads();                       // all waves done reading buf before reuse
  }

  const int sel = lane >> 4, nl = lane & 15;
  #pragma unroll
  for (int im = 0; im < 2; ++im) {
    #pragma unroll
    for (int in = 0; in < 2; ++in) {
      const v8f accA = im ? (in ? acc11 : acc10) : (in ? acc01 : acc00);
      const v8f accU = im ? (in ? accg11 : accg10) : (in ? accg01 : accg00);
      const int n = n0 + waveN * 32 + in * 16 + nl;
      #pragma unroll
      for (int r = 0; r < 8; ++r) {
        const int m = m0 + waveM * 32 + im * 16 + r + 8 * sel;
        const size_t idx = (size_t)m * N + n;
        if (MODE == 1) {
          ((float*)outp)[idx] = resid[idx] + accA[r];
        } else if (MODE == 2) {
          const float g = accA[r], u = accU[r];
          ((unsigned short*)outp)[idx] = f2bf((g / (1.0f + __expf(-g))) * u);
        } else if (MODE == 3) {            // Vt[b][n][s], s-padded to 384
          const int bb = m / 361, s = m - bb * 361;
          ((unsigned short*)outp)[((size_t)bb * 512 + n) * 384 + s] = f2bf(accA[r]);
        } else {                           // MODE 4: fused RoPE, bf16 store
          const float val = accA[r];
          const float par = __shfl_xor(val, 1, 32);   // column n^1 lives in lane^1
          const int s = m % 361, d = n & 63;
          const float cs = ropeC[s * 64 + d], sn = ropeS[s * 64 + d];
          const float o = (d & 1) ? (par * sn - val * cs) : (val * cs + par * sn);
          ((unsigned short*)outp)[idx] = f2bf(o);
        }
      }
    }
  }
}

// ---------------------------------------------------------------------------
// Attention: per (b, h, 16-query tile). 384-key padding; async LDS staging.
// Pass1: scores = Q K^T * 0.125 (dual-accumulator WMMA) -> fp32 LDS (pad=-1e30)
// Softmax in LDS -> bf16 P (pad cols -> exact 0).  Pass2: out = P @ Vt (WMMA).
// LDS: sQ 2KB + sKV 16KB + sS 24KB + sP 12KB = 54KB
// ---------------------------------------------------------------------------
#define ATT_TQ 16
#define ATT_TK 128
#define ATT_SP 384

__global__ __launch_bounds__(256)
void attention_kernel(const unsigned short* __restrict__ Q,
                      const unsigned short* __restrict__ K,
                      const unsigned short* __restrict__ Vt,
                      unsigned short* __restrict__ O) {
  __shared__ __align__(16) unsigned short sQ[ATT_TQ * 64];
  __shared__ __align__(16) unsigned short sKV[ATT_TK * 64];
  __shared__ __align__(16) float          sS[ATT_TQ * ATT_SP];
  __shared__ __align__(16) unsigned short sP[ATT_TQ * ATT_SP];

  const int t = threadIdx.x, lane = t & 31, wid = t >> 5;
  const int qt = blockIdx.x;               // 0..22
  const int bh = blockIdx.y;               // 0..1023
  const int b = bh >> 3, h = bh & 7;
  const size_t baseTok = (size_t)b * 361;
  const int col0 = h * 64;
  const int nl = lane & 15, sel = lane >> 4;

  const unsigned long long qBase = (unsigned long long)(uintptr_t)Q;
  const unsigned long long kBase = (unsigned long long)(uintptr_t)K;
  const unsigned long long vBase = (unsigned long long)(uintptr_t)Vt;

  // --- Q tile: one async b64 per thread (16 rows x 64 halfs) ---
  {
    const int qr = t >> 4, c = (t & 15) * 4;
    const unsigned goff = (unsigned)(((baseTok + qt * ATT_TQ + qr) * 512 + col0 + c) * 2);
    async_g2l_b64(lds_addr(&sQ[qr * 64 + c]), goff, qBase);
  }

  // staging geometry shared by both passes
  const int ky = t >> 1, dpart = (t & 1) * 32;           // pass 1 (K rows)
  const unsigned lK = lds_addr(&sKV[ky * 64 + dpart]);
  const unsigned gK0 = (unsigned)(((baseTok + ky) * 512 + col0 + dpart) * 2);
  const int vd = t >> 2, vkc = (t & 3) * 32;             // pass 2 (Vt rows)
  const unsigned lV = lds_addr(&sKV[vd * ATT_TK + vkc]);
  const unsigned gV0 = (unsigned)((((size_t)(b * 512 + col0 + vd)) * 384 + vkc) * 2);

  // --- pass 1: scores ---
  #pragma unroll
  for (int kt = 0; kt < 3; ++kt) {
    __syncthreads();                       // previous tile fully consumed
    const unsigned gk = gK0 + (unsigned)kt * (ATT_TK * 512 * 2);
    async_g2l_b128o<0>(lK, gk, kBase);
    async_g2l_b128o<16>(lK, gk, kBase);
    async_g2l_b128o<32>(lK, gk, kBase);
    async_g2l_b128o<48>(lK, gk, kBase);
    wait_asynccnt<0>();
    __syncthreads();

    const int kwl = wid * 16;
    // two independent accumulator chains (no WMMA->WMMA RAW hazard)
    v8f sc0 = {}, sc1 = {};
    {
      const v16bf a  = load_a_frag(&sQ[0], 64, lane);
      const v16bf bb = load_b_frag(&sKV[kwl * 64 + 0], 64, lane);
      sc0 = wmma_bf16(a, bb, sc0);
    }
    {
      const v16bf a  = load_a_frag(&sQ[32], 64, lane);
      const v16bf bb = load_b_frag(&sKV[kwl * 64 + 32], 64, lane);
      sc1 = wmma_bf16(a, bb, sc1);
    }
    const int gcol = kt * ATT_TK + kwl + nl;
    #pragma unroll
    for (int r = 0; r < 8; ++r) {
      const int row = r + 8 * sel;
      sS[row * ATT_SP + kt * ATT_TK + kwl + nl] =
          (gcol < 361) ? (sc0[r] + sc1[r]) * 0.125f : -1.0e30f;
    }
  }
  __syncthreads();

  // --- softmax (2 rows per wave) ---
  #pragma unroll
  for (int rr = 0; rr < 2; ++rr) {
    const int row = wid * 2 + rr;
    float mx = -1.0e30f;
    for (int i = lane; i < ATT_SP; i += 32) mx = fmaxf(mx, sS[row * ATT_SP + i]);
    #pragma unroll
    for (int o = 16; o > 0; o >>= 1) mx = fmaxf(mx, __shfl_xor(mx, o, 32));
    float sm = 0.f;
    for (int i = lane; i < ATT_SP; i += 32) sm += __expf(sS[row * ATT_SP + i] - mx);
    #pragma unroll
    for (int o = 16; o > 0; o >>= 1) sm += __shfl_xor(sm, o, 32);
    const float inv = 1.0f / sm;
    for (int i = lane; i < ATT_SP; i += 32)
      sP[row * ATT_SP + i] = f2bf(__expf(sS[row * ATT_SP + i] - mx) * inv);
  }

  // --- pass 2: out = P @ V (V already [d][s] transposed, 384-padded) ---
  v8f o0 = {}, o1 = {};
  #pragma unroll
  for (int kt = 0; kt < 3; ++kt) {
    __syncthreads();
    const unsigned gv = gV0 + (unsigned)kt * (ATT_TK * 2);
    async_g2l_b128o<0>(lV, gv, vBase);
    async_g2l_b128o<16>(lV, gv, vBase);
    async_g2l_b128o<32>(lV, gv, vBase);
    async_g2l_b128o<48>(lV, gv, vBase);
    wait_asynccnt<0>();
    __syncthreads();
    if (wid < 4) {
      const int dw = wid * 16;
      #pragma unroll
      for (int ks = 0; ks < ATT_TK; ks += 64) {   // alternate chains o0 / o1
        {
          const v16bf a  = load_a_frag(&sP[kt * ATT_TK + ks], ATT_SP, lane);
          const v16bf bb = load_b_frag(&sKV[dw * ATT_TK + ks], ATT_TK, lane);
          o0 = wmma_bf16(a, bb, o0);
        }
        {
          const v16bf a  = load_a_frag(&sP[kt * ATT_TK + ks + 32], ATT_SP, lane);
          const v16bf bb = load_b_frag(&sKV[dw * ATT_TK + ks + 32], ATT_TK, lane);
          o1 = wmma_bf16(a, bb, o1);
        }
      }
    }
  }

  if (wid < 4) {
    const int dw = wid * 16;
    #pragma unroll
    for (int r = 0; r < 8; ++r) {
      const int q = qt * ATT_TQ + r + 8 * sel;
      if (q < 361)
        O[(baseTok + q) * 512 + col0 + dw + nl] = f2bf(o0[r] + o1[r]);
    }
  }
}

// ---------------------------------------------------------------------------
// Host: pipeline
// ---------------------------------------------------------------------------
extern "C" void kernel_launch(void* const* d_in, const int* in_sizes, int n_in,
                              void* d_out, int out_size, void* d_ws, size_t ws_size,
                              hipStream_t stream) {
  const float* x     = (const float*)d_in[0];
  const float* Wq    = (const float*)d_in[1];
  const float* Wk    = (const float*)d_in[2];
  const float* Wv    = (const float*)d_in[3];
  const float* Wo    = (const float*)d_in[4];
  const float* Wg    = (const float*)d_in[5];
  const float* Wu    = (const float*)d_in[6];
  const float* Wd    = (const float*)d_in[7];
  const float* g_in  = (const float*)d_in[8];
  const float* g_out = (const float*)d_in[9];

  const int M = 46208;         // B * S = 128 * 361 (divisible by 128)
  const int D = 512, F = 1024;

  char* ws = (char*)d_ws;
  size_t off = 0;
  auto alloc = [&](size_t bytes) -> void* {
    void* p = ws + off;
    off += (bytes + 255) & ~(size_t)255;
    return p;
  };
  unsigned short* WqT = (unsigned short*)alloc((size_t)D * D * 2);
  unsigned short* WkT = (unsigned short*)alloc((size_t)D * D * 2);
  unsigned short* WvT = (unsigned short*)alloc((size_t)D * D * 2);
  unsigned short* WoT = (unsigned short*)alloc((size_t)D * D * 2);
  unsigned short* WgT = (unsigned short*)alloc((size_t)D * F * 2);
  unsigned short* WuT = (unsigned short*)alloc((size_t)D * F * 2);
  unsigned short* WdT = (unsigned short*)alloc((size_t)F * D * 2);
  float* ropeC = (float*)alloc((size_t)361 * 64 * 4);
  float* ropeS = (float*)alloc((size_t)361 * 64 * 4);
  unsigned short* hbf = (unsigned short*)alloc((size_t)M * D * 2);
  unsigned short* Qb  = (unsigned short*)alloc((size_t)M * D * 2);
  unsigned short* Kb  = (unsigned short*)alloc((size_t)M * D * 2);  // contiguous after Qb
  unsigned short* Vt  = (unsigned short*)alloc((size_t)128 * 512 * 384 * 2); // [b][d512][384]
  unsigned short* Ob  = (unsigned short*)alloc((size_t)M * D * 2);
  unsigned short* GU  = Qb;    // reuse Q+K regions (M x 1024 bf16) in FFN stage

  // 1) weights -> bf16, transposed to [N][K]
  convert_transpose_kernel<<<(D * D + 255) / 256, 256, 0, stream>>>(Wq, WqT, D, D);
  convert_transpose_kernel<<<(D * D + 255) / 256, 256, 0, stream>>>(Wk, WkT, D, D);
  convert_transpose_kernel<<<(D * D + 255) / 256, 256, 0, stream>>>(Wv, WvT, D, D);
  convert_transpose_kernel<<<(D * D + 255) / 256, 256, 0, stream>>>(Wo, WoT, D, D);
  convert_transpose_kernel<<<(D * F + 255) / 256, 256, 0, stream>>>(Wg, WgT, D, F);
  convert_transpose_kernel<<<(D * F + 255) / 256, 256, 0, stream>>>(Wu, WuT, D, F);
  convert_transpose_kernel<<<(F * D + 255) / 256, 256, 0, stream>>>(Wd, WdT, F, D);

  // 2) RoPE tables
  rope_table_kernel<<<(361 * 64 + 255) / 256, 256, 0, stream>>>(ropeC, ropeS);

  // 3) h = rmsnorm(x, g_in) -> bf16
  rmsnorm_kernel<<<M / 8, 256, 0, stream>>>(x, g_in, hbf, M);

  // 4) projections (async double-buffered WMMA GEMM); RoPE fused into Q/K epilogue
  gemm128_kernel<4><<<dim3(D / 64, M / 128), 256, 0, stream>>>(hbf, WqT, nullptr, nullptr, ropeC, ropeS, Qb, M, D, D);
  gemm128_kernel<4><<<dim3(D / 64, M / 128), 256, 0, stream>>>(hbf, WkT, nullptr, nullptr, ropeC, ropeS, Kb, M, D, D);
  gemm128_kernel<3><<<dim3(D / 64, M / 128), 256, 0, stream>>>(hbf, WvT, nullptr, nullptr, nullptr, nullptr, Vt, M, D, D);

  // 5) attention
  attention_kernel<<<dim3(23, 128 * 8), 256, 0, stream>>>(Qb, Kb, Vt, Ob);

  // 6) x1 = x + attn_out @ Wo  (fp32 into d_out)
  gemm128_kernel<1><<<dim3(D / 64, M / 128), 256, 0, stream>>>(Ob, WoT, nullptr, x, nullptr, nullptr, d_out, M, D, D);

  // 7) h2 = rmsnorm(x1, g_out) -> bf16 (reuse hbf)
  rmsnorm_kernel<<<M / 8, 256, 0, stream>>>((const float*)d_out, g_out, hbf, M);

  // 8) gu = silu(h2 @ Wg) * (h2 @ Wu) -> bf16 (dual-B fused GEMM)
  gemm128_kernel<2><<<dim3(F / 64, M / 128), 256, 0, stream>>>(hbf, WgT, WuT, nullptr, nullptr, nullptr, GU, M, F, D);

  // 9) out = x1 + gu @ Wd  (in-place fp32 residual add on d_out)
  gemm128_kernel<1><<<dim3(D / 64, M / 128), 256, 0, stream>>>(GU, WdT, nullptr, (const float*)d_out, nullptr, nullptr, d_out, M, D, F);
}
